// ScaledDotProductAttention_39530878992675
// MI455X (gfx1250) — matrix operand order repaired and checked by reading, not compiled, not run on hardware
//
#include <hip/hip_runtime.h>
#include <hip/hip_bf16.h>
#include <cstdint>

typedef __attribute__((ext_vector_type(16))) _Float16 v16h;
typedef __attribute__((ext_vector_type(8)))  _Float16 v8h;
typedef __attribute__((ext_vector_type(4)))  _Float16 v4h;
typedef __attribute__((ext_vector_type(8)))  float    v8f;
typedef __attribute__((ext_vector_type(4)))  float    v4f;

#define Bc 4
#define Hc 16
#define Sc 2048
#define Dc 64
#define SSTRIDE 2052   // fp32 score row stride in LDS dwords (2052%64==4 -> bank skew)
#define PSTRIDE 2056   // f16 prob row stride in halves (4112B row, 16B aligned, 4-bank skew)
#define VCHUNK  256    // K-rows of V staged per iteration
#define VTS     264    // Vt row stride in halves (528B row: 16B aligned, 4-bank skew)

#define LDS_SCORES_BYTES (16 * SSTRIDE * 4)          // 131328 (aliased by Vt in phase 3)
#define LDS_P_BYTES      (16 * PSTRIDE * 2)          // 65792
#define LDS_ORED_BYTES   (4 * 256 * 4)               // 4096
#define LDS_TOTAL (LDS_SCORES_BYTES + LDS_P_BYTES + LDS_ORED_BYTES)

__global__ __launch_bounds__(256)
void sdpa_wmma_kernel(const float* __restrict__ q,
                      const float* __restrict__ k,
                      const float* __restrict__ v,
                      const uint8_t* __restrict__ mask,
                      float* __restrict__ out,
                      float* __restrict__ attn)
{
    extern __shared__ char smem[];
    float*    S_sc = (float*)smem;                               // [16][SSTRIDE] fp32 scores
    _Float16* Vt   = (_Float16*)smem;                            // [64][VTS] f16 V^T (aliases S_sc, phase 3)
    _Float16* Pbuf = (_Float16*)(smem + LDS_SCORES_BYTES);       // [16][PSTRIDE] f16 probs
    float*    Ored = (float*)(smem + LDS_SCORES_BYTES + LDS_P_BYTES); // [4][16][16] partial out

    const int tid   = threadIdx.x;
    const int wid   = tid >> 5;
    const int lane  = tid & 31;
    const int lhalf = lane >> 4;   // 0 | 1 (lane half)
    const int l16   = lane & 15;

    const int qt = blockIdx.x & 127;      // q-row tile (128 per head)
    const int bh = blockIdx.x >> 7;       // b*H + h
    const int b  = bh >> 4;

    const float*   qbase = q + ((size_t)bh * Sc + qt * 16) * Dc;
    const float*   kbase = k + (size_t)bh * Sc * Dc;
    const float*   vbase = v + (size_t)bh * Sc * Dc;
    const uint8_t* mbase = mask + ((size_t)b * Sc + qt * 16) * (size_t)Sc;

    // ---------------- Phase 1: S = mask( (Q/8) K^T ) into LDS ----------------
    // A layout (16x32 f16): lane<16: M=l16, halves 0..15 -> K=c*32+0..15
    //                       lane>=16: same M, K range +16.
    v16h a0, a1;
    {
        const float* qr = qbase + (size_t)l16 * Dc;
        #pragma unroll
        for (int c = 0; c < 2; ++c) {
            const int base1 = 32 * c + lhalf * 16;
            const int base2 = 32 * c + lhalf * 16 + 8;
            v16h a;
            #pragma unroll
            for (int j = 0; j < 8; ++j) {
                a[j]     = (_Float16)(0.125f * qr[base1 + j]);
                a[j + 8] = (_Float16)(0.125f * qr[base2 + j]);
            }
            if (c == 0) a0 = a; else a1 = a;
        }
    }

    for (int t = 0; t < 16; ++t) {
        const int nt   = wid + t * 8;       // column tile index (0..127)
        const int col0 = nt * 16;

        // Prefetch next K tile row for this lane into cache (global_prefetch_b8)
        if (t < 15)
            __builtin_prefetch(kbase + (size_t)(col0 + 128 + l16) * Dc + lhalf * 16, 0, 1);

        // B fragments: lane = column N=l16; K-range d = c*32 + lhalf*16 + j (16 contig floats)
        v16h b0, b1;
        {
            const float* kr = kbase + (size_t)(col0 + l16) * Dc + lhalf * 16;
            #pragma unroll
            for (int j = 0; j < 16; ++j) b0[j] = (_Float16)kr[j];
            #pragma unroll
            for (int j = 0; j < 16; ++j) b1[j] = (_Float16)kr[32 + j];
        }

        v8f acc = {};
        acc = __builtin_amdgcn_wmma_f32_16x16x32_f16(false, a0, false, b0,
                                                     (short)0, acc, false, false);
        acc = __builtin_amdgcn_wmma_f32_16x16x32_f16(false, a1, false, b1,
                                                     (short)0, acc, false, false);

        // C layout: acc[r] is (M = r + 8*lhalf, N = l16)
        const int n = col0 + l16;
        #pragma unroll
        for (int r = 0; r < 8; ++r) {
            const int m = r + 8 * lhalf;
            const float sv = mbase[(size_t)m * Sc + n] ? -1e9f : acc[r];
            S_sc[m * SSTRIDE + n] = sv;
        }
    }
    __syncthreads();

    // ---------------- Phase 2: row softmax; stream attn (NT) + f16 P ----------------
    {
        const int row = tid >> 4;
        const int j4  = (tid & 15) * 4;
        float* srow = S_sc + row * SSTRIDE;

        float mmax = -3.4e38f;
        for (int c = j4; c < Sc; c += 64) {
            const v4f sv = *(const v4f*)(srow + c);
            mmax = fmaxf(fmaxf(fmaxf(mmax, sv[0]), sv[1]), fmaxf(sv[2], sv[3]));
        }
        #pragma unroll
        for (int o = 8; o; o >>= 1) mmax = fmaxf(mmax, __shfl_xor(mmax, o, 16));

        float ssum = 0.0f;
        for (int c = j4; c < Sc; c += 64) {
            v4f sv = *(const v4f*)(srow + c);
            v4f e;
            #pragma unroll
            for (int i = 0; i < 4; ++i) e[i] = __expf(sv[i] - mmax);
            *(v4f*)(srow + c) = e;
            ssum += (e[0] + e[1]) + (e[2] + e[3]);
        }
        #pragma unroll
        for (int o = 8; o; o >>= 1) ssum += __shfl_xor(ssum, o, 16);
        const float inv = 1.0f / ssum;

        float*    arow = attn + ((size_t)bh * Sc + qt * 16 + row) * (size_t)Sc;
        _Float16* prow = Pbuf + row * PSTRIDE;
        for (int c = j4; c < Sc; c += 64) {
            const v4f e = *(const v4f*)(srow + c);
            const v4f p = e * inv;
            __builtin_nontemporal_store(p, (v4f*)(arow + c));  // 1.07GB stream: bypass temporal caching
            v4h ph;
            #pragma unroll
            for (int i = 0; i < 4; ++i) ph[i] = (_Float16)p[i];
            *(v4h*)(prow + c) = ph;
        }
    }
    __syncthreads();

    // ---------------- Phase 3: out = P @ V via staged V^T chunks ----------------
    // 8 waves: dtile = wid&3 (16-wide d slice), khalf = wid>>2 (half of each 256-K chunk)
    const int dtile = wid & 3;
    const int khalf = wid >> 2;
    const int rg_lo = tid >> 4;   // staging row-group base (0..15)
    const int dg    = tid & 15;   // staging d-group (0..15)

    v8f oacc = {};
    const _Float16* prow = Pbuf + (size_t)l16 * PSTRIDE;

    for (int chunk = 0; chunk < Sc / VCHUNK; ++chunk) {
        const int kc0 = chunk * VCHUNK;

        // --- Stage: cooperative coalesced load of V[kc0..kc0+256), transpose to Vt[d][k] f16 ---
        // Each thread: 4 tiles of 4 rows x 4 d; float4 loads, ds_store_b64 columns.
        #pragma unroll
        for (int i = 0; i < 4; ++i) {
            const int rg = rg_lo + 16 * i;              // 0..63
            const int r0 = kc0 + rg * 4;
            const int d0 = dg * 4;
            v4f rowv[4];
            #pragma unroll
            for (int rr = 0; rr < 4; ++rr)
                rowv[rr] = *(const v4f*)(vbase + (size_t)(r0 + rr) * Dc + d0);
            #pragma unroll
            for (int c = 0; c < 4; ++c) {
                v4h colh;
                #pragma unroll
                for (int rr = 0; rr < 4; ++rr) colh[rr] = (_Float16)rowv[rr][c];
                *(v4h*)(Vt + (size_t)(d0 + c) * VTS + rg * 4) = colh;
            }
        }
        __syncthreads();

        // --- Compute: 4 WMMAs per wave over this chunk ---
        const int kb0 = khalf * (VCHUNK / 2);
        #pragma unroll
        for (int kb = 0; kb < VCHUNK / 2; kb += 32) {
            const int kloc  = kb0 + kb;          // within chunk
            const int kglob = kc0 + kloc;

            // A fragment from Pbuf (two 16B ds loads)
            const v8h alo = *(const v8h*)(prow + kglob + lhalf * 16);
            const v8h ahi = *(const v8h*)(prow + kglob + lhalf * 16 + 8);
            const v16h a = __builtin_shufflevector(alo, ahi,
                            0,1,2,3,4,5,6,7,8,9,10,11,12,13,14,15);

            // B fragment from Vt: lane N=l16 -> d, K = kloc + lhalf*16 + j (16 contig halves)
            const _Float16* vtr = Vt + (size_t)(dtile * 16 + l16) * VTS + kloc + lhalf * 16;
            const v8h blo = *(const v8h*)(vtr);
            const v8h bhi = *(const v8h*)(vtr + 8);
            const v16h bb = __builtin_shufflevector(blo, bhi,
                            0,1,2,3,4,5,6,7,8,9,10,11,12,13,14,15);

            oacc = __builtin_amdgcn_wmma_f32_16x16x32_f16(false, a, false, bb,
                                                          (short)0, oacc, false, false);
        }
        __syncthreads();
    }

    // --- Split-K reduction across the two khalf waves, then coalesced store ---
    if (khalf == 1) {
        #pragma unroll
        for (int r = 0; r < 8; ++r) {
            const int m = r + 8 * lhalf;
            Ored[dtile * 256 + m * 16 + l16] = oacc[r];
        }
    }
    __syncthreads();
    if (khalf == 0) {
        float* obase = out + ((size_t)bh * Sc + qt * 16) * Dc + dtile * 16 + l16;
        #pragma unroll
        for (int r = 0; r < 8; ++r) {
            const int m = r + 8 * lhalf;
            const float ov = oacc[r] + Ored[dtile * 256 + m * 16 + l16];
            __builtin_nontemporal_store(ov, obase + (size_t)m * Dc);
        }
    }
}

extern "C" void kernel_launch(void* const* d_in, const int* in_sizes, int n_in,
                              void* d_out, int out_size, void* d_ws, size_t ws_size,
                              hipStream_t stream) {
    const float*   q    = (const float*)d_in[0];
    const float*   k    = (const float*)d_in[1];
    const float*   v    = (const float*)d_in[2];
    const uint8_t* mask = (const uint8_t*)d_in[3];

    float* out  = (float*)d_out;                                   // [B,H,S,D]
    float* attn = out + (size_t)Bc * Hc * Sc * Dc;                 // [B,H,S,S]

    (void)in_sizes; (void)n_in; (void)out_size; (void)d_ws; (void)ws_size;

    hipFuncSetAttribute((const void*)sdpa_wmma_kernel,
                        hipFuncAttributeMaxDynamicSharedMemorySize, LDS_TOTAL);

    const int grid = Bc * Hc * (Sc / 16);   // 8192 workgroups
    sdpa_wmma_kernel<<<dim3(grid), dim3(256), LDS_TOTAL, stream>>>(q, k, v, mask, out, attn);
}
